// EdgeConvModel_4767413699081
// MI455X (gfx1250) — matrix operand order repaired and checked by reading, not compiled, not run on hardware
//
#include <hip/hip_runtime.h>

// ---------------------------------------------------------------------------
// EdgeConv GNN for MI455X (gfx1250).
//  - per-edge MLPs on v_wmma_f32_16x16x32_f16 (f16 in, f32 accumulate)
//  - weights staged in LDS (f16, column-major, zero-padded) -> ds_load_b128
//  - segment_max: float atomicMax into an f32 buffer memset to 0xFF (-NaN
//    sentinel, absorbed by maximumNumber); tiny per-node fixup kernel then
//    writes an f16 gather copy (isolated nodes -> 0), so edge gathers are
//    pure 16B vector loads + v_pk_sub_f16 (no decode VALU).
// ---------------------------------------------------------------------------

#define NN_C 100000
#define NE_C 1600000

typedef __attribute__((ext_vector_type(16))) _Float16 v16h;
typedef __attribute__((ext_vector_type(8)))  _Float16 v8h;
typedef __attribute__((ext_vector_type(8)))  float    v8f;

// Compiler-level ordering between LDS store phase and LDS reload phase.
// Per-wave DS ops are in-order in hardware; the wait is belt & braces.
#define WAVE_LDS_SYNC() do { \
    __builtin_amdgcn_wave_barrier(); \
    asm volatile("s_wait_dscnt 0" ::: "memory"); \
  } while (0)

// Convert f32 row-major [K][N] weight to f16 col-major padded [Npad][Kpad].
__device__ __forceinline__ void fill_w(_Float16* dst, const float* __restrict__ w,
                                       int K, int N, int Kpad, int Npad) {
  for (int i = threadIdx.x; i < Npad * Kpad; i += blockDim.x) {
    int n = i / Kpad, k = i - n * Kpad;
    dst[i] = (k < K && n < N) ? (_Float16)w[(size_t)k * N + n] : (_Float16)0.0f;
  }
}

// One dense layer: C[nt] = A x W(+bias). B-fragment: two runs of 8 halves
// per lane from col-major LDS weights (k = kt*32 + (h/8)*16 + hi*8 + h%8, n = nt*16+col).
template<int KT, int NT>
__device__ __forceinline__ void run_layer(const v16h* A, const _Float16* W, int Kpad,
                                          const float* __restrict__ bias, int H,
                                          v8f* C, int col, int hi) {
#pragma unroll
  for (int nt = 0; nt < NT; ++nt) {
    const int n = nt * 16 + col;
    const float bv = (n < H) ? bias[n] : 0.0f;
    v8f c;
#pragma unroll
    for (int i = 0; i < 8; ++i) c[i] = bv;
#pragma unroll
    for (int kt = 0; kt < KT; ++kt) {
      const v8h bl = *(const v8h*)(W + (size_t)n * Kpad + kt * 32 + hi * 8);
      const v8h bh = *(const v8h*)(W + (size_t)n * Kpad + kt * 32 + 16 + hi * 8);
      const v16h b = __builtin_shufflevector(bl, bh, 0, 1, 2, 3, 4, 5, 6, 7, 8, 9, 10, 11, 12, 13, 14, 15);
      c = __builtin_amdgcn_wmma_f32_16x16x32_f16(false, A[kt], false, b,
                                                 (short)0, c, false, false);
    }
    C[nt] = c;
  }
}

template<int NT>
__device__ __forceinline__ void lrelu(v8f* C) {
#pragma unroll
  for (int nt = 0; nt < NT; ++nt)
#pragma unroll
    for (int i = 0; i < 8; ++i)
      C[nt][i] = fmaxf(C[nt][i], 0.1f * C[nt][i]);   // leaky(0.1) == max(x,0.1x)
}

// C-frag (row m = r+8*hi, col n = nt*16+col) -> per-wave LDS act [16][AW] f16.
template<int NT>
__device__ __forceinline__ void stage(_Float16* act, int AW, const v8f* C,
                                      int col, int hi) {
#pragma unroll
  for (int nt = 0; nt < NT; ++nt)
#pragma unroll
    for (int r = 0; r < 8; ++r)
      act[(r + 8 * hi) * AW + nt * 16 + col] = (_Float16)C[nt][r];
}

// Rebuild A-frags from staged act: lane row m=col, K-runs of 8 contiguous f16.
// Runs with base >= NPprev are hard zeros (K padding).
template<int KT>
__device__ __forceinline__ void loadA(const _Float16* act, int AW, int NPprev,
                                      v16h* A, int col, int hi) {
#pragma unroll
  for (int kt = 0; kt < KT; ++kt) {
    v16h a;
#pragma unroll
    for (int run = 0; run < 2; ++run) {
      const int base = kt * 32 + run * 16 + hi * 8;
      v8h t = {};
      if (base < NPprev) t = *(const v8h*)(act + col * AW + base);
#pragma unroll
      for (int j = 0; j < 8; ++j) a[run * 8 + j] = t[j];
    }
    A[kt] = a;
  }
}

// ---------------------------------------------------------------------------
// BatchNorm stats (biased variance) + scale/shift finalize
// ---------------------------------------------------------------------------
__global__ void col_stats(const float* __restrict__ data, long long rows, int cols,
                          float* __restrict__ sums) {
  __shared__ float ls[64];
  if (threadIdx.x < 2 * cols) ls[threadIdx.x] = 0.0f;
  __syncthreads();
  const long long total = rows * (long long)cols;
  for (long long i = blockIdx.x * (long long)blockDim.x + threadIdx.x;
       i < total; i += (long long)gridDim.x * blockDim.x) {
    float v = data[i];
    int c = (int)(i % cols);
    atomicAdd(&ls[c], v);
    atomicAdd(&ls[cols + c], v * v);
  }
  __syncthreads();
  if (threadIdx.x < 2 * cols) atomicAdd(&sums[threadIdx.x], ls[threadIdx.x]);
}

__global__ void finalize_stats(const float* __restrict__ sums, float invRows, int cols,
                               const float* __restrict__ g, const float* __restrict__ b,
                               float* __restrict__ scale, float* __restrict__ shift) {
  int c = threadIdx.x;
  if (c < cols) {
    float m = sums[c] * invRows;
    float v = sums[cols + c] * invRows - m * m;
    float s = g[c] * rsqrtf(v + 1e-5f);
    scale[c] = s;
    shift[c] = b[c] - m * s;
  }
}

// Per-node fixup: NaN sentinel (no incoming edge) -> 0, write f16 gather copy.
__global__ void fixup_to_h(const float* __restrict__ g, _Float16* __restrict__ h,
                           int total) {
  for (int i = blockIdx.x * blockDim.x + threadIdx.x; i < total;
       i += gridDim.x * blockDim.x) {
    float v = g[i];
    h[i] = (_Float16)((v == v) ? v : 0.0f);
  }
}

// ---------------------------------------------------------------------------
// EdgeConv layer: msg = MLP3([xi, xj-xi]); gmax[dst] = fmax(gmax[dst], msg).
// RAW=false: input is f16 node features (pure vector-load gather).
// RAW=true : input is raw f32 x with fused BatchNorm affine.
// ---------------------------------------------------------------------------
template<int F_IN, int H1, int H2, int H3, bool RAW>
__global__ __launch_bounds__(128) void edgeconv_kernel(
    const float* __restrict__ xraw, const _Float16* __restrict__ xh,
    const float* __restrict__ scl, const float* __restrict__ shf,
    const int* __restrict__ srcI, const int* __restrict__ dstI,
    const float* __restrict__ w1, const float* __restrict__ b1,
    const float* __restrict__ w2, const float* __restrict__ b2,
    const float* __restrict__ w3, const float* __restrict__ b3,
    float* __restrict__ gmax, int nEdges) {
  constexpr int K1  = 2 * F_IN;
  constexpr int K1P = (K1 + 31) & ~31;  constexpr int KT1 = K1P / 32;
  constexpr int NP1 = (H1 + 15) & ~15;  constexpr int NT1 = NP1 / 16;
  constexpr int K2P = (H1 + 31) & ~31;  constexpr int KT2 = K2P / 32;
  constexpr int NP2 = (H2 + 15) & ~15;  constexpr int NT2 = NP2 / 16;
  constexpr int K3P = (H2 + 31) & ~31;  constexpr int KT3 = K3P / 32;
  constexpr int NP3 = (H3 + 15) & ~15;  constexpr int NT3 = NP3 / 16;
  constexpr int AW  = (K2P > K3P) ? K2P : K3P;
  constexpr int KTMAX = KT1 > KT2 ? (KT1 > KT3 ? KT1 : KT3)
                                  : (KT2 > KT3 ? KT2 : KT3);
  constexpr int WAVES = 4;

  __shared__ __align__(32) _Float16 sW1[NP1 * K1P];
  __shared__ __align__(32) _Float16 sW2[NP2 * K2P];
  __shared__ __align__(32) _Float16 sW3[NP3 * K3P];
  __shared__ __align__(32) _Float16 sAct[WAVES * 16 * AW];

  fill_w(sW1, w1, K1, H1, K1P, NP1);
  fill_w(sW2, w2, H1, H2, K2P, NP2);
  fill_w(sW3, w3, H2, H3, K3P, NP3);
  __syncthreads();

  const int lane = threadIdx.x & 31;
  const int col  = lane & 15;
  const int hi   = lane >> 4;
  const int wIn  = threadIdx.x >> 5;
  _Float16* act  = sAct + wIn * 16 * AW;

  const int nTiles = nEdges >> 4;
  const int waveId = blockIdx.x * WAVES + wIn;
  const int nWaves = gridDim.x * WAVES;

  for (int tile = waveId; tile < nTiles; tile += nWaves) {
    const int e0  = tile * 16;
    const int myE = e0 + col;
    const int s   = srcI[myE];
    const int d   = dstI[myE];

    if (tile + nWaves < nTiles) {                      // global_prefetch_b8
      __builtin_prefetch(srcI + (tile + nWaves) * 16, 0, 0);
      __builtin_prefetch(dstI + (tile + nWaves) * 16, 0, 0);
    }

    // ---- Build layer-1 A fragments: [xi | xj - xi], K padded with zeros ----
    v16h A[KTMAX];
#pragma unroll
    for (int kt = 0; kt < KT1; ++kt) {
      v16h a;
#pragma unroll
      for (int run = 0; run < 2; ++run) {
        const int base = kt * 32 + run * 16 + hi * 8;  // runs never straddle regions
        v8h t = {};
        if (base < F_IN) {
          if constexpr (RAW) {
#pragma unroll
            for (int j = 0; j < 8; ++j)
              t[j] = (_Float16)(xraw[(long long)d * F_IN + base + j] * scl[base + j]
                                + shf[base + j]);
          } else {
            t = *(const v8h*)(xh + (long long)d * F_IN + base);
          }
        } else if (base < 2 * F_IN) {
          const int k2 = base - F_IN;
          if constexpr (RAW) {
#pragma unroll
            for (int j = 0; j < 8; ++j)
              t[j] = (_Float16)((xraw[(long long)s * F_IN + k2 + j]
                                 - xraw[(long long)d * F_IN + k2 + j]) * scl[k2 + j]);
          } else {
            const v8h ts = *(const v8h*)(xh + (long long)s * F_IN + k2);
            const v8h td = *(const v8h*)(xh + (long long)d * F_IN + k2);
            t = ts - td;                               // v_pk_sub_f16
          }
        }
#pragma unroll
        for (int j = 0; j < 8; ++j) a[run * 8 + j] = t[j];
      }
      A[kt] = a;
    }

    v8f C[NT1];
    run_layer<KT1, NT1>(A, sW1, K1P, b1, H1, C, col, hi);
    lrelu<NT1>(C);
    stage<NT1>(act, AW, C, col, hi);
    WAVE_LDS_SYNC();
    loadA<KT2>(act, AW, NP1, A, col, hi);

    run_layer<KT2, NT2>(A, sW2, K2P, b2, H2, C, col, hi);
    lrelu<NT2>(C);
    stage<NT2>(act, AW, C, col, hi);
    WAVE_LDS_SYNC();
    loadA<KT3>(act, AW, NP2, A, col, hi);

    run_layer<KT3, NT3>(A, sW3, K3P, b3, H3, C, col, hi);

    // ---- segment_max onto dst: native float max atomic ----
#pragma unroll
    for (int nt = 0; nt < NT3; ++nt) {
      const int n = nt * 16 + col;
      if (n < H3) {
#pragma unroll
        for (int r = 0; r < 8; ++r) {
          const int eg = e0 + r + 8 * hi;
          const int dd = dstI[eg];
          atomicMax(gmax + (size_t)dd * H3 + n, C[nt][r]);
        }
      }
    }
  }
}

// ---------------------------------------------------------------------------
// Edge MLP: [x[src](64) | x[dst](64) | bn(e)(10)] -> 64->32->16->8->2
// ---------------------------------------------------------------------------
__global__ __launch_bounds__(128) void edge_mlp_kernel(
    const _Float16* __restrict__ xh, const float* __restrict__ eat,
    const float* __restrict__ sclE, const float* __restrict__ shfE,
    const int* __restrict__ srcI, const int* __restrict__ dstI,
    const float* __restrict__ w1, const float* __restrict__ b1,
    const float* __restrict__ w2, const float* __restrict__ b2,
    const float* __restrict__ w3, const float* __restrict__ b3,
    const float* __restrict__ w4, const float* __restrict__ b4,
    const float* __restrict__ w5, const float* __restrict__ b5,
    float* __restrict__ outp, int nEdges) {
  constexpr int K1P = 160, KT1 = 5;                 // 138 padded to 160
  constexpr int NP1 = 64,  NT1 = 4,  K2P = 64, KT2 = 2;
  constexpr int NP2 = 32,  NT2 = 2,  K3P = 32, KT3 = 1;
  constexpr int NP3 = 16,  NT3 = 1,  K4P = 32, KT4 = 1;
  constexpr int NP4 = 16,             K5P = 32, KT5 = 1;
  constexpr int AW = 64, WAVES = 4;

  __shared__ __align__(32) _Float16 sW1[NP1 * K1P];
  __shared__ __align__(32) _Float16 sW2[NP2 * K2P];
  __shared__ __align__(32) _Float16 sW3[NP3 * K3P];
  __shared__ __align__(32) _Float16 sW4[NP4 * K4P];
  __shared__ __align__(32) _Float16 sW5[NP4 * K5P];
  __shared__ __align__(32) _Float16 sAct[WAVES * 16 * AW];

  fill_w(sW1, w1, 138, 64, K1P, NP1);
  fill_w(sW2, w2, 64, 32, K2P, NP2);
  fill_w(sW3, w3, 32, 16, K3P, NP3);
  fill_w(sW4, w4, 16, 8,  K4P, NP4);
  fill_w(sW5, w5, 8,  2,  K5P, NP4);
  __syncthreads();

  const int lane = threadIdx.x & 31;
  const int col  = lane & 15;
  const int hi   = lane >> 4;
  const int wIn  = threadIdx.x >> 5;
  _Float16* act  = sAct + wIn * 16 * AW;

  const int nTiles = nEdges >> 4;
  const int waveId = blockIdx.x * WAVES + wIn;
  const int nWaves = gridDim.x * WAVES;

  for (int tile = waveId; tile < nTiles; tile += nWaves) {
    const int e0  = tile * 16;
    const int myE = e0 + col;
    const int s   = srcI[myE];
    const int d   = dstI[myE];

    if (tile + nWaves < nTiles) {
      __builtin_prefetch(eat + (long long)(tile + nWaves) * 160, 0, 0);
    }

    v16h A[KT1];
#pragma unroll
    for (int kt = 0; kt < KT1; ++kt) {
      v16h a;
#pragma unroll
      for (int run = 0; run < 2; ++run) {
        const int base = kt * 32 + run * 16 + hi * 8;
        v8h t = {};
        if (base < 64) {
          t = *(const v8h*)(xh + (long long)s * 64 + base);
        } else if (base < 128) {
          t = *(const v8h*)(xh + (long long)d * 64 + (base - 64));
        } else {
#pragma unroll
          for (int j = 0; j < 8; ++j) {                // e region: per-elem guard
            const int k2 = base - 128 + j;
            if (k2 < 10)
              t[j] = (_Float16)(eat[(long long)myE * 10 + k2] * sclE[k2] + shfE[k2]);
          }
        }
#pragma unroll
        for (int j = 0; j < 8; ++j) a[run * 8 + j] = t[j];
      }
      A[kt] = a;
    }

    v8f C[NT1];
    run_layer<KT1, NT1>(A, sW1, K1P, b1, 64, C, col, hi);
    lrelu<NT1>(C);
    stage<NT1>(act, AW, C, col, hi);
    WAVE_LDS_SYNC();
    loadA<KT2>(act, AW, NP1, A, col, hi);

    run_layer<KT2, NT2>(A, sW2, K2P, b2, 32, C, col, hi);
    lrelu<NT2>(C);
    stage<NT2>(act, AW, C, col, hi);
    WAVE_LDS_SYNC();
    loadA<KT3>(act, AW, NP2, A, col, hi);

    run_layer<KT3, NT3>(A, sW3, K3P, b3, 16, C, col, hi);
    lrelu<NT3>(C);
    stage<NT3>(act, AW, C, col, hi);
    WAVE_LDS_SYNC();
    loadA<KT4>(act, AW, NP3, A, col, hi);

    run_layer<KT4, 1>(A, sW4, K4P, b4, 8, C, col, hi);
    lrelu<1>(C);
    stage<1>(act, AW, C, col, hi);
    WAVE_LDS_SYNC();
    loadA<KT5>(act, AW, NP4, A, col, hi);

    run_layer<KT5, 1>(A, sW5, K5P, b5, 2, C, col, hi);

    if (col < 2) {
#pragma unroll
      for (int r = 0; r < 8; ++r) {
        const int eg = e0 + r + 8 * hi;
        outp[(long long)eg * 2 + col] = C[0][r];
      }
    }
  }
}

// ---------------------------------------------------------------------------
// Host side
// ---------------------------------------------------------------------------
extern "C" void kernel_launch(void* const* d_in, const int* in_sizes, int n_in,
                              void* d_out, int out_size, void* d_ws, size_t ws_size,
                              hipStream_t stream) {
  const int NN = NN_C, NE = NE_C;
  auto F = [&](int i) { return (const float*)d_in[i]; };

  // --- detect top-level ordering: insertion (x,ei,e,xb,params) vs sorted ---
  int ix, iei, ie, p;
  if (in_sizes[0] == NN * 16) { ix = 0; iei = 1; ie = 2; p = 4; }
  else                        { ie = 0; iei = 1; p = 2; ix = n_in - 2; }

  const float* x   = F(ix);
  const float* eat = F(ie);
  const int* ei    = (const int*)d_in[iei];
  const int* srcI  = ei;
  const int* dstI  = ei + NE;

  // --- detect params ordering: insertion (bn_node first, 16) vs sorted (bn_edge, 10)
  const float *bnNg, *bnNb, *bnEg, *bnEb;
  const float *W[14], *B[14];  // 0-2 nn0, 3-5 nn1, 6-8 nn2, 9-13 edge_mlp
  if (in_sizes[p] == 16) {
    bnNg = F(p); bnNb = F(p + 1); bnEg = F(p + 2); bnEb = F(p + 3); p += 4;
    for (int i = 0; i < 14; ++i) { W[i] = F(p++); B[i] = F(p++); }
  } else {
    bnEg = F(p); bnEb = F(p + 1); bnNg = F(p + 2); bnNb = F(p + 3); p += 4;
    for (int i = 9; i < 14; ++i) { W[i] = F(p++); B[i] = F(p++); }  // edge_mlp
    for (int i = 0; i < 9;  ++i) { W[i] = F(p++); B[i] = F(p++); }  // nn0,nn1,nn2
  }

  // --- workspace layout ---
  char* w8 = (char*)d_ws;
  float* sumsX = (float*)(w8 + 0);      // 32 f
  float* sumsE = (float*)(w8 + 128);    // 20 f
  float* sclX  = (float*)(w8 + 256);  float* shfX = sclX + 16;
  float* sclE  = (float*)(w8 + 384);  float* shfE = sclE + 16;
  float*    gmax = (float*)(w8 + 512);                     // [NN,64] f32, reused
  _Float16* h0 = (_Float16*)(w8 + 512 + (size_t)NN * 64 * 4);  // [NN,24] f16
  _Float16* h1 = h0 + (size_t)NN * 24;                         // [NN,32] f16
  _Float16* h2 = h1 + (size_t)NN * 32;                         // [NN,64] f16

  hipMemsetAsync(d_ws, 0, 512, stream);             // zero BN accumulators

  // --- BN stats (biased variance) -> scale/shift ---
  col_stats<<<256, 256, 0, stream>>>(x, NN, 16, sumsX);
  col_stats<<<1024, 256, 0, stream>>>(eat, NE, 10, sumsE);
  finalize_stats<<<1, 32, 0, stream>>>(sumsX, 1.0f / NN, 16, bnNg, bnNb, sclX, shfX);
  finalize_stats<<<1, 32, 0, stream>>>(sumsE, 1.0f / NE, 10, bnEg, bnEb, sclE, shfE);

  const int GRID = 1024, BLK = 128;

  // --- EdgeConv stack: memset(-NaN) -> conv (atomic fmax) -> fixup (f16) ---
  hipMemsetAsync(gmax, 0xFF, (size_t)NN * 24 * 4, stream);
  edgeconv_kernel<16, 48, 24, 24, true><<<GRID, BLK, 0, stream>>>(
      x, nullptr, sclX, shfX, srcI, dstI,
      W[0], B[0], W[1], B[1], W[2], B[2], gmax, NE);
  fixup_to_h<<<1024, 256, 0, stream>>>(gmax, h0, NN * 24);

  hipMemsetAsync(gmax, 0xFF, (size_t)NN * 32 * 4, stream);
  edgeconv_kernel<24, 64, 32, 32, false><<<GRID, BLK, 0, stream>>>(
      nullptr, h0, nullptr, nullptr, srcI, dstI,
      W[3], B[3], W[4], B[4], W[5], B[5], gmax, NE);
  fixup_to_h<<<1024, 256, 0, stream>>>(gmax, h1, NN * 32);

  hipMemsetAsync(gmax, 0xFF, (size_t)NN * 64 * 4, stream);
  edgeconv_kernel<32, 128, 64, 64, false><<<GRID, BLK, 0, stream>>>(
      nullptr, h1, nullptr, nullptr, srcI, dstI,
      W[6], B[6], W[7], B[7], W[8], B[8], gmax, NE);
  fixup_to_h<<<1024, 256, 0, stream>>>(gmax, h2, NN * 64);

  // --- MetaLayer edge model ---
  edge_mlp_kernel<<<GRID, BLK, 0, stream>>>(
      h2, eat, sclE, shfE, srcI, dstI,
      W[9], B[9], W[10], B[10], W[11], B[11], W[12], B[12], W[13], B[13],
      (float*)d_out, NE);
  (void)n_in; (void)out_size; (void)ws_size;
}
